// Self_Attn_57286273794058
// MI455X (gfx1250) — matrix operand order repaired and checked by reading, not compiled, not run on hardware
//
#include <hip/hip_runtime.h>
#include <hip/hip_bf16.h>
#include <math.h>

#define B_  4
#define C_  256
#define H_  128
#define W_  128
#define H2_ 64
#define W2_ 64
#define N_  4096   // H2_*W2_

typedef _Float16 v16h __attribute__((ext_vector_type(16)));
typedef _Float16 v8h  __attribute__((ext_vector_type(8)));
typedef float    v8f  __attribute__((ext_vector_type(8)));

// ---------------------------------------------------------------------------
// WMMA fragment loaders (wave32, 16x16x32 f16 shapes, per CDNA5 ISA)
//
// A matrix 16x32 (MxK): lane L in [0,16) holds row M=L, K in {0..7, 16..23};
// lane L in [16,32) holds row M=L-16, K in {8..15, 24..31}.
// Per lane: two contiguous 8-half (16B) chunks at rowPtr + kb and rowPtr+kb+16,
// kb = (lane>>4)*8.
static __device__ __forceinline__ v16h load_a_frag(const _Float16* rowPtr, int lane) {
  const int kb = (lane >> 4) * 8;
  v8h lo = *(const v8h*)(rowPtr + kb);
  v8h hi = *(const v8h*)(rowPtr + kb + 16);
  v16h r;
#pragma unroll
  for (int e = 0; e < 8; ++e) { r[e] = lo[e]; r[e + 8] = hi[e]; }
  return r;
}

// B matrix 32x16 (KxN): lane L in [0,16) holds col N=L, K in {0..7, 16..23};
// lane L in [16,32) holds col N=L-16, K in {8..15, 24..31}. Per lane: two 8-half
// chunks at colPtr + (lane>>4)*8 and +16, where colPtr walks K contiguously for that column.
static __device__ __forceinline__ v16h load_b_frag(const _Float16* colPtr, int lane) {
  const _Float16* p = colPtr + (lane >> 4) * 8;
  v8h lo = *(const v8h*)(p);
  v8h hi = *(const v8h*)(p + 16);
  v16h r;
#pragma unroll
  for (int e = 0; e < 8; ++e) { r[e] = lo[e]; r[e + 8] = hi[e]; }
  return r;
}

static __device__ __forceinline__ v8f wmma_f16(v16h a, v16h b, v8f c) {
  return __builtin_amdgcn_wmma_f32_16x16x32_f16(false, a, false, b, (short)0, c, false, false);
}

// ---------------------------------------------------------------------------
// 1. f32 -> f16 convert (weights)
__global__ void cvt_f16_kernel(const float* __restrict__ src, _Float16* __restrict__ dst, int n) {
  int i = blockIdx.x * blockDim.x + threadIdx.x;
  if (i < n) dst[i] = (_Float16)src[i];
}

// 2. 2x2 mean pool of t,b,m -> transposed (B, N, C) f16 (contiguous in C per pixel)
__global__ void pool_kernel(const float* __restrict__ t, const float* __restrict__ bsrc,
                            const float* __restrict__ m,
                            _Float16* __restrict__ tpT, _Float16* __restrict__ bpT,
                            _Float16* __restrict__ mpT) {
  size_t idx = (size_t)blockIdx.x * blockDim.x + threadIdx.x;
  if (idx >= (size_t)B_ * N_ * C_) return;
  int c = (int)(idx % C_);
  int i = (int)((idx / C_) % N_);
  int b = (int)(idx / ((size_t)C_ * N_));
  int y = i / W2_, x = i % W2_;
  size_t base = (((size_t)b * C_ + c) * H_ + 2 * y) * W_ + 2 * x;
  size_t o = ((size_t)b * N_ + i) * C_ + c;
  tpT[o] = (_Float16)(0.25f * (t[base] + t[base + 1] + t[base + W_] + t[base + W_ + 1]));
  bpT[o] = (_Float16)(0.25f * (bsrc[base] + bsrc[base + 1] + bsrc[base + W_] + bsrc[base + W_ + 1]));
  mpT[o] = (_Float16)(0.25f * (m[base] + m[base + 1] + m[base + W_] + m[base + W_ + 1]));
}

// 3. 1x1 conv as GEMM: X[o,i] = sum_c W[o,c]*P[b,i,c] + bias[o]
//    W: (C,C) f16 row-major; P: (B,N,C) f16. 32x32 macro-tile per wave (2x2 WMMA).
//    transposeOut=1 -> write (B,N,C) f16 (for q,k); 0 -> write (B,C,N) f16 (for v).
__global__ void conv1x1_wmma_kernel(const _Float16* __restrict__ Wh,
                                    const float* __restrict__ bias,
                                    const _Float16* __restrict__ P,
                                    _Float16* __restrict__ out, int transposeOut) {
  const int lane = threadIdx.x & 31;
  int T = blockIdx.x * 4 + (threadIdx.x >> 5);       // macro-tile index (wave-uniform)
  const int tilesO = C_ / 32, tilesI = N_ / 32;
  if (T >= B_ * tilesO * tilesI) return;
  int b  = T / (tilesO * tilesI);
  int r  = T % (tilesO * tilesI);
  int o0 = (r / tilesI) * 32;
  int i0 = (r % tilesI) * 32;

  v8f acc[2][2];
#pragma unroll
  for (int ti = 0; ti < 2; ++ti)
#pragma unroll
    for (int tj = 0; tj < 2; ++tj) acc[ti][tj] = (v8f){0.f,0.f,0.f,0.f,0.f,0.f,0.f,0.f};

  const _Float16* aRow0 = Wh + (size_t)(o0 + (lane & 15)) * C_;
  const _Float16* aRow1 = aRow0 + (size_t)16 * C_;
  const _Float16* bCol0 = P + ((size_t)b * N_ + i0 + (lane & 15)) * C_;
  const _Float16* bCol1 = bCol0 + (size_t)16 * C_;
#pragma unroll 2
  for (int kk = 0; kk < C_; kk += 32) {
    v16h a0 = load_a_frag(aRow0 + kk, lane);
    v16h a1 = load_a_frag(aRow1 + kk, lane);
    v16h b0 = load_b_frag(bCol0 + kk, lane);
    v16h b1 = load_b_frag(bCol1 + kk, lane);
    acc[0][0] = wmma_f16(a0, b0, acc[0][0]);
    acc[0][1] = wmma_f16(a0, b1, acc[0][1]);
    acc[1][0] = wmma_f16(a1, b0, acc[1][0]);
    acc[1][1] = wmma_f16(a1, b1, acc[1][1]);
  }
  // C/D layout: lane n = lane&15, rows m = q + 8*(lane>>4)
  const int n = lane & 15;
  const int mb = 8 * (lane >> 4);
#pragma unroll
  for (int ti = 0; ti < 2; ++ti) {
#pragma unroll
    for (int tj = 0; tj < 2; ++tj) {
      int ob = o0 + ti * 16 + mb;
      int ib = i0 + tj * 16 + n;
      if (transposeOut) {
        _Float16* dst = out + ((size_t)b * N_ + ib) * C_ + ob;
        v8h st;
#pragma unroll
        for (int q = 0; q < 8; ++q) st[q] = (_Float16)(acc[ti][tj][q] + bias[ob + q]);
        *(v8h*)dst = st;
      } else {
#pragma unroll
        for (int q = 0; q < 8; ++q)
          out[((size_t)b * C_ + ob + q) * N_ + ib] = (_Float16)(acc[ti][tj][q] + bias[ob + q]);
      }
    }
  }
}

// 4. energy[i,j] = sum_c qT[b,i,c]*kT[b,j,c]   (one batch; E is N_ x N_ f32)
//    32x32 macro-tile per wave.
__global__ void energy_wmma_kernel(const _Float16* __restrict__ qT,
                                   const _Float16* __restrict__ kT,
                                   float* __restrict__ E, int b) {
  const int lane = threadIdx.x & 31;
  int T = blockIdx.x * 4 + (threadIdx.x >> 5);
  const int tiles = N_ / 32;                          // 128
  if (T >= tiles * tiles) return;
  int i0 = (T / tiles) * 32;
  int j0 = (T % tiles) * 32;

  v8f acc[2][2];
#pragma unroll
  for (int ti = 0; ti < 2; ++ti)
#pragma unroll
    for (int tj = 0; tj < 2; ++tj) acc[ti][tj] = (v8f){0.f,0.f,0.f,0.f,0.f,0.f,0.f,0.f};

  const _Float16* aRow0 = qT + ((size_t)b * N_ + i0 + (lane & 15)) * C_;
  const _Float16* aRow1 = aRow0 + (size_t)16 * C_;
  const _Float16* bCol0 = kT + ((size_t)b * N_ + j0 + (lane & 15)) * C_;
  const _Float16* bCol1 = bCol0 + (size_t)16 * C_;
#pragma unroll 2
  for (int kk = 0; kk < C_; kk += 32) {
    v16h a0 = load_a_frag(aRow0 + kk, lane);
    v16h a1 = load_a_frag(aRow1 + kk, lane);
    v16h b0 = load_b_frag(bCol0 + kk, lane);
    v16h b1 = load_b_frag(bCol1 + kk, lane);
    acc[0][0] = wmma_f16(a0, b0, acc[0][0]);
    acc[0][1] = wmma_f16(a0, b1, acc[0][1]);
    acc[1][0] = wmma_f16(a1, b0, acc[1][0]);
    acc[1][1] = wmma_f16(a1, b1, acc[1][1]);
  }
  const int n = lane & 15;
  const int mb = 8 * (lane >> 4);
#pragma unroll
  for (int ti = 0; ti < 2; ++ti)
#pragma unroll
    for (int tj = 0; tj < 2; ++tj)
#pragma unroll
      for (int q = 0; q < 8; ++q)
        E[(size_t)(i0 + ti * 16 + mb + q) * N_ + j0 + tj * 16 + n] = acc[ti][tj][q];
}

// 5. row softmax over f32 energy -> f16 attn (one batch, block per row)
__global__ void softmax_kernel(const float* __restrict__ E, _Float16* __restrict__ A) {
  __shared__ float red[256];
  const int i = blockIdx.x;
  const float* row = E + (size_t)i * N_;
  float mx = -INFINITY;
  for (int j = threadIdx.x; j < N_; j += 256) mx = fmaxf(mx, row[j]);
  red[threadIdx.x] = mx; __syncthreads();
  for (int s = 128; s > 0; s >>= 1) {
    if (threadIdx.x < s) red[threadIdx.x] = fmaxf(red[threadIdx.x], red[threadIdx.x + s]);
    __syncthreads();
  }
  mx = red[0]; __syncthreads();
  float sum = 0.f;
  for (int j = threadIdx.x; j < N_; j += 256) sum += __expf(row[j] - mx);
  red[threadIdx.x] = sum; __syncthreads();
  for (int s = 128; s > 0; s >>= 1) {
    if (threadIdx.x < s) red[threadIdx.x] += red[threadIdx.x + s];
    __syncthreads();
  }
  const float inv = 1.0f / red[0];
  _Float16* arow = A + (size_t)i * N_;
  for (int j = threadIdx.x; j < N_; j += 256)
    arow[j] = (_Float16)(__expf(row[j] - mx) * inv);
}

// 6. out[c,i] = sum_j v[b,c,j]*attn[i,j]  (one batch; out f32 (B,C,N))
//    32x32 macro-tile per wave.
__global__ void outgemm_wmma_kernel(const _Float16* __restrict__ V,
                                    const _Float16* __restrict__ A,
                                    float* __restrict__ O, int b) {
  const int lane = threadIdx.x & 31;
  int T = blockIdx.x * 4 + (threadIdx.x >> 5);
  const int tilesC = C_ / 32, tilesI = N_ / 32;       // 8 x 128
  if (T >= tilesC * tilesI) return;
  int c0 = (T / tilesI) * 32;
  int i0 = (T % tilesI) * 32;

  v8f acc[2][2];
#pragma unroll
  for (int ti = 0; ti < 2; ++ti)
#pragma unroll
    for (int tj = 0; tj < 2; ++tj) acc[ti][tj] = (v8f){0.f,0.f,0.f,0.f,0.f,0.f,0.f,0.f};

  const _Float16* aRow0 = V + ((size_t)b * C_ + c0 + (lane & 15)) * N_;  // contiguous in j
  const _Float16* aRow1 = aRow0 + (size_t)16 * N_;
  const _Float16* bCol0 = A + (size_t)(i0 + (lane & 15)) * N_;           // attn row, contiguous in j
  const _Float16* bCol1 = bCol0 + (size_t)16 * N_;
#pragma unroll 2
  for (int jj = 0; jj < N_; jj += 32) {
    __builtin_prefetch(aRow0 + jj + 512, 0, 0);   // global_prefetch_b8
    v16h a0 = load_a_frag(aRow0 + jj, lane);
    v16h a1 = load_a_frag(aRow1 + jj, lane);
    v16h b0 = load_b_frag(bCol0 + jj, lane);
    v16h b1 = load_b_frag(bCol1 + jj, lane);
    acc[0][0] = wmma_f16(a0, b0, acc[0][0]);
    acc[0][1] = wmma_f16(a0, b1, acc[0][1]);
    acc[1][0] = wmma_f16(a1, b0, acc[1][0]);
    acc[1][1] = wmma_f16(a1, b1, acc[1][1]);
  }
  const int n = lane & 15;
  const int mb = 8 * (lane >> 4);
#pragma unroll
  for (int ti = 0; ti < 2; ++ti)
#pragma unroll
    for (int tj = 0; tj < 2; ++tj)
#pragma unroll
      for (int q = 0; q < 8; ++q)
        O[((size_t)b * C_ + c0 + ti * 16 + mb + q) * N_ + i0 + tj * 16 + n] = acc[ti][tj][q];
}

// 7. bilinear 2x upsample (half-pixel, edge clamp) + (t+b+m) residual
__global__ void upsample_add_kernel(const float* __restrict__ O,
                                    const float* __restrict__ t,
                                    const float* __restrict__ bsrc,
                                    const float* __restrict__ m,
                                    float* __restrict__ out) {
  size_t idx = (size_t)blockIdx.x * blockDim.x + threadIdx.x;
  if (idx >= (size_t)B_ * C_ * H_ * W_) return;
  int x = (int)(idx % W_);
  int y = (int)((idx / W_) % H_);
  size_t bc = idx / ((size_t)H_ * W_);
  float sy = y * 0.5f - 0.25f;
  float sx = x * 0.5f - 0.25f;
  int y0 = (int)floorf(sy), x0 = (int)floorf(sx);
  float wy = sy - (float)y0, wx = sx - (float)x0;
  int y1 = min(y0 + 1, H2_ - 1), x1 = min(x0 + 1, W2_ - 1);
  y0 = max(0, min(y0, H2_ - 1));
  x0 = max(0, min(x0, W2_ - 1));
  const float* src = O + bc * (size_t)(H2_ * W2_);
  float v00 = src[(size_t)y0 * W2_ + x0];
  float v01 = src[(size_t)y0 * W2_ + x1];
  float v10 = src[(size_t)y1 * W2_ + x0];
  float v11 = src[(size_t)y1 * W2_ + x1];
  float up = (1.f - wy) * ((1.f - wx) * v00 + wx * v01)
           +        wy  * ((1.f - wx) * v10 + wx * v11);
  out[idx] = up + t[idx] + bsrc[idx] + m[idx];
}

// ---------------------------------------------------------------------------
extern "C" void kernel_launch(void* const* d_in, const int* in_sizes, int n_in,
                              void* d_out, int out_size, void* d_ws, size_t ws_size,
                              hipStream_t stream) {
  (void)in_sizes; (void)n_in; (void)out_size; (void)ws_size;
  const float* t  = (const float*)d_in[0];
  const float* b  = (const float*)d_in[1];
  const float* m  = (const float*)d_in[2];
  const float* Wq = (const float*)d_in[3];
  const float* bq = (const float*)d_in[4];
  const float* Wk = (const float*)d_in[5];
  const float* bk = (const float*)d_in[6];
  const float* Wv = (const float*)d_in[7];
  const float* bv = (const float*)d_in[8];
  float* out = (float*)d_out;

  // workspace carve-out (all offsets 256B aligned)
  char* ws = (char*)d_ws;
  const size_t szPooled = (size_t)B_ * N_ * C_ * sizeof(_Float16); // 8 MB
  const size_t szW      = (size_t)C_ * C_ * sizeof(_Float16);      // 128 KB
  _Float16* tpT  = (_Float16*)(ws);                       // (B,N,C)
  _Float16* bpT  = (_Float16*)(ws + szPooled);
  _Float16* mpT  = (_Float16*)(ws + 2 * szPooled);
  _Float16* qT   = (_Float16*)(ws + 3 * szPooled);        // (B,N,C)
  _Float16* kT   = (_Float16*)(ws + 4 * szPooled);        // (B,N,C)
  _Float16* vH   = (_Float16*)(ws + 5 * szPooled);        // (B,C,N)
  _Float16* WqH  = (_Float16*)(ws + 6 * szPooled);
  _Float16* WkH  = (_Float16*)(ws + 6 * szPooled + szW);
  _Float16* WvH  = (_Float16*)(ws + 6 * szPooled + 2 * szW);
  char* p = ws + 6 * szPooled + 3 * szW;
  float*    energy = (float*)p;                           // N x N f32 (one batch, reused)
  p += (size_t)N_ * N_ * sizeof(float);
  _Float16* attnH  = (_Float16*)p;                        // N x N f16 (one batch, reused)
  p += (size_t)N_ * N_ * sizeof(_Float16);
  float*    obuf   = (float*)p;                           // (B,C,N) f32

  // 1. convert weights
  cvt_f16_kernel<<<(C_ * C_ + 255) / 256, 256, 0, stream>>>(Wq, WqH, C_ * C_);
  cvt_f16_kernel<<<(C_ * C_ + 255) / 256, 256, 0, stream>>>(Wk, WkH, C_ * C_);
  cvt_f16_kernel<<<(C_ * C_ + 255) / 256, 256, 0, stream>>>(Wv, WvH, C_ * C_);

  // 2. pool
  {
    size_t tot = (size_t)B_ * N_ * C_;
    pool_kernel<<<(int)((tot + 255) / 256), 256, 0, stream>>>(t, b, m, tpT, bpT, mpT);
  }

  // 3. q/k/v GEMMs (q,k transposed-out for energy A/B layouts; v row-major in j)
  {
    int waves = B_ * (C_ / 32) * (N_ / 32);     // 4096 macro-tiles
    int blocks = waves / 4;
    conv1x1_wmma_kernel<<<blocks, 128, 0, stream>>>(WqH, bq, tpT, qT, 1);
    conv1x1_wmma_kernel<<<blocks, 128, 0, stream>>>(WkH, bk, bpT, kT, 1);
    conv1x1_wmma_kernel<<<blocks, 128, 0, stream>>>(WvH, bv, mpT, vH, 0);
  }

  // 4-6. per-batch attention (energy buffer fits in L2; batches sequential)
  {
    int etiles  = (N_ / 32) * (N_ / 32);        // 16384 macro-tiles
    int eblocks = etiles / 4;
    int otiles  = (C_ / 32) * (N_ / 32);        // 1024 macro-tiles
    int oblocks = otiles / 4;
    for (int bb = 0; bb < B_; ++bb) {
      energy_wmma_kernel<<<eblocks, 128, 0, stream>>>(qT, kT, energy, bb);
      softmax_kernel<<<N_, 256, 0, stream>>>(energy, attnH);
      outgemm_wmma_kernel<<<oblocks, 128, 0, stream>>>(vH, attnH, obuf, bb);
    }
  }

  // 7. upsample + residual
  {
    size_t tot = (size_t)B_ * C_ * H_ * W_;
    upsample_add_kernel<<<(int)((tot + 255) / 256), 256, 0, stream>>>(obuf, t, b, m, out);
  }
}
